// ODELayer_28793460752530
// MI455X (gfx1250) — matrix operand order repaired and checked by reading, not compiled, not run on hardware
//
#include <hip/hip_runtime.h>

// ODE fused Euler solver for MI455X (gfx1250, wave32, WMMA).
// z:[16384,256] f32, W1:[256,256], W2:[256,256], 25 Euler steps.
// Rows of z are independent across steps -> one fully fused kernel.
// Transposed formulation: weights are WMMA A-operands (pre-packed bf16 frags),
// activations are B-operands / f32 D-accumulators. z state lives in LDS in a
// B-fragment-friendly layout. A-fragment loads are software-pipelined two
// deep so load latency overlaps WMMA execution.

#define BN 16384
#define DN 256
#define HN 256
#define NSTEPS 25

typedef __attribute__((ext_vector_type(16))) __bf16 v16bf;
typedef __attribute__((ext_vector_type(8)))  __bf16 v8bf;
typedef __attribute__((ext_vector_type(8)))  float  v8f;

// ---------------------------------------------------------------------------
// Pack W1^T and W2^T (bf16) into WMMA A-fragment order used here:
//   half index o = ((mt*8+kt)*32 + lane)*16 + i
//   A row  m = mt*16 + (lane&15)
//   A col  k = kt*32 + (lane>>4)*16 + i   (contiguous k order, matches B)
// GEMM1 A = W1^T (A[h][d] = W1[d][h]); GEMM2 A = W2^T (A[d][h] = W2[h][d]).
// Both sources are [256,256] row-major, so value = W[k*256 + m].
// ---------------------------------------------------------------------------
__global__ __launch_bounds__(256) void pack_weights_kernel(
    const float* __restrict__ W1, const float* __restrict__ W2,
    __bf16* __restrict__ wsW) {
  int idx   = blockIdx.x * blockDim.x + threadIdx.x;   // 0 .. 131071
  int which = idx >> 16;                               // 0 = W1, 1 = W2
  int o     = idx & 65535;
  int frag  = o >> 9;
  int lane  = (o >> 4) & 31;
  int i     = o & 15;
  int mt    = frag >> 3;
  int kt    = frag & 7;
  int m     = mt * 16 + (lane & 15);
  int k     = kt * 32 + ((lane >> 4) << 4) + i;
  const float* W = which ? W2 : W1;
  wsW[idx] = (__bf16)W[k * 256 + m];
}

// Load fragment #idx (idx = mt*8+kt) from a lane-major packed frag buffer
// (32B per lane). Works for LDS (W1) and global (W2) pointers.
__device__ __forceinline__ v16bf load_afrag_idx(const __bf16* sW, int idx,
                                                int lane) {
  const __bf16* p = sW + (idx << 9) + (lane << 4);
  v8bf lo = *(const v8bf*)p;
  v8bf hi = *(const v8bf*)(p + 8);
  v16bf a;
#pragma unroll
  for (int r = 0; r < 8; ++r) { a[r] = lo[r]; a[8 + r] = hi[r]; }
  return a;
}

// Convert two stacked f32 16x16 D-tiles (rows 0-15 / 16-31 of a 32x16 K-slab)
// into one bf16 32x16 B-fragment. B layout: lanes 0-15 hold K=0-15, lanes
// 16-31 hold K=16-31; a lane-XOR-16 shuffle supplies the missing 8 rows.
__device__ __forceinline__ v16bf make_bfrag(const v8f t0, const v8f t1,
                                            int hh) {
  v16bf b;
#pragma unroll
  for (int r = 0; r < 8; ++r) {
    float a0 = t0[r], a1 = t1[r];
    float o0 = __shfl_xor(a0, 16, 32);
    float o1 = __shfl_xor(a1, 16, 32);
    b[r]     = (__bf16)(hh ? o1 : a0);   // K = (hh?16:0) + r
    b[8 + r] = (__bf16)(hh ? a1 : o0);   // K = (hh?16:0) + 8 + r
  }
  return b;
}

// ---------------------------------------------------------------------------
// Fused solver. Block = 256 threads = 8 waves; each wave owns 16 batch cols.
// LDS: W1 frags 128KB + z tiles 8x16KB = 128KB + biases 3KB = 259KB dynamic.
// z^T LDS layout (per wave, f32): idx(f, n) = (f/16)*256 + n*16 + (f%16)
//   -> a 32-row B-frag slab p is 16 contiguous f32 per lane at
//      ((2p + lane/16)*16 + lane%16)*16, and a GEMM2 D-tile mt updates 8
//      contiguous f32 per lane at (mt*16 + lane%16)*16 + 8*(lane/16).
// ---------------------------------------------------------------------------
__global__ __launch_bounds__(256) void ode_fused_kernel(
    const float* __restrict__ z0, const float* __restrict__ trange,
    const float* __restrict__ b1, const float* __restrict__ wt,
    const float* __restrict__ b2, const __bf16* __restrict__ wsW,
    float* __restrict__ zout) {
  extern __shared__ char smem[];
  __bf16* sW1 = (__bf16*)smem;                       // 131072 B
  float*  zS  = (float*)(smem + 131072);             // 131072 B (8 waves)
  float*  sB1 = (float*)(smem + 262144);             // 1KB
  float*  sWt = sB1 + 256;
  float*  sB2 = sB1 + 512;

  // Stage W1 fragments + biases into LDS.
  {
    const uint4* src = (const uint4*)wsW;            // first 8192 uint4 = W1
    uint4*       dst = (uint4*)smem;
    for (int q = threadIdx.x; q < 8192; q += 256) dst[q] = src[q];
    if (threadIdx.x < 256) {
      sB1[threadIdx.x] = b1[threadIdx.x];
      sWt[threadIdx.x] = wt[threadIdx.x];
      sB2[threadIdx.x] = b2[threadIdx.x];
    }
  }
  __syncthreads();

  const __bf16* wsW2 = wsW + 65536;  // W2 frags stay in global (L2-resident)

  const int lane = threadIdx.x & 31;
  const int hh   = lane >> 4;        // lane half
  const int nn   = lane & 15;        // batch column within wave tile
  const int wid  = threadIdx.x >> 5;
  const int j    = (blockIdx.x * 8 + wid) * 16 + nn;  // global batch row
  float* zw = zS + wid * 4096;       // this wave's 16KB z tile

  // Load z0 tile into LDS (frag-friendly layout). 64B contiguous per lane.
#pragma unroll
  for (int p = 0; p < 8; ++p) {
    const float4* gsrc = (const float4*)(z0 + (size_t)j * DN + p * 32 + hh * 16);
    float4* ldst = (float4*)(zw + ((p * 2 + hh) * 16 + nn) * 16);
#pragma unroll
    for (int q = 0; q < 4; ++q) ldst[q] = gsrc[q];
  }

  const float dt = trange[1] - trange[0];

#pragma unroll 1
  for (int s = 0; s < NSTEPS; ++s) {
    // Compiler fence: keeps fragment loads inside the loop (no LICM -> no
    // spills) and orders cross-lane LDS z traffic between steps.
    asm volatile("" ::: "memory");
    const float t = trange[s];

    // z^T -> bf16 B-fragments (8 K-slabs of 32), straight from LDS.
    v16bf zB[8];
#pragma unroll
    for (int p = 0; p < 8; ++p) {
      const float4* zp = (const float4*)(zw + ((p * 2 + hh) * 16 + nn) * 16);
      float4 f0 = zp[0], f1 = zp[1], f2 = zp[2], f3 = zp[3];
      v16bf zb;
      zb[0]  = (__bf16)f0.x; zb[1]  = (__bf16)f0.y;
      zb[2]  = (__bf16)f0.z; zb[3]  = (__bf16)f0.w;
      zb[4]  = (__bf16)f1.x; zb[5]  = (__bf16)f1.y;
      zb[6]  = (__bf16)f1.z; zb[7]  = (__bf16)f1.w;
      zb[8]  = (__bf16)f2.x; zb[9]  = (__bf16)f2.y;
      zb[10] = (__bf16)f2.z; zb[11] = (__bf16)f2.w;
      zb[12] = (__bf16)f3.x; zb[13] = (__bf16)f3.y;
      zb[14] = (__bf16)f3.z; zb[15] = (__bf16)f3.w;
      zB[p] = zb;
    }

    // GEMM1: h^T = tanh(W1^T * z^T + b1 + t*wt), emitted directly as B-frags.
    // A-fragment stream software-pipelined 2 deep (LDS).
    v16bf hB[8];
    {
      v16bf a0 = load_afrag_idx(sW1, 0, lane);
      v16bf a1 = load_afrag_idx(sW1, 1, lane);
#pragma unroll
      for (int p = 0; p < 8; ++p) {
        v8f hpair[2];
#pragma unroll
        for (int q = 0; q < 2; ++q) {
          const int mt = 2 * p + q;
          v8f acc = {};
#pragma unroll
          for (int kt = 0; kt < 8; ++kt) {
            const int nidx = mt * 8 + kt + 2;
            v16bf an = (nidx < 128) ? load_afrag_idx(sW1, nidx, lane) : a0;
            acc = __builtin_amdgcn_wmma_f32_16x16x32_bf16(
                false, a0, false, zB[kt], (short)0, acc, false, false);
            a0 = a1; a1 = an;
          }
          const int base = mt * 16 + hh * 8;
#pragma unroll
          for (int r = 0; r < 8; ++r)
            acc[r] = tanhf(acc[r] + sB1[base + r] + t * sWt[base + r]);
          hpair[q] = acc;
        }
        hB[p] = make_bfrag(hpair[0], hpair[1], hh);
      }
    }

    // GEMM2: z^T += dt * (W2^T * h^T + b2); z updated in place in LDS.
    // A-fragment stream software-pipelined 2 deep (global, L2-hot).
    {
      v16bf a0 = load_afrag_idx(wsW2, 0, lane);
      v16bf a1 = load_afrag_idx(wsW2, 1, lane);
#pragma unroll
      for (int mt = 0; mt < 16; ++mt) {
        v8f g = {};
#pragma unroll
        for (int kt = 0; kt < 8; ++kt) {
          const int nidx = mt * 8 + kt + 2;
          v16bf an = (nidx < 128) ? load_afrag_idx(wsW2, nidx, lane) : a0;
          g = __builtin_amdgcn_wmma_f32_16x16x32_bf16(
              false, a0, false, hB[kt], (short)0, g, false, false);
          a0 = a1; a1 = an;
        }
        const int base = mt * 16 + hh * 8;
        float4* zp = (float4*)(zw + (mt * 16 + nn) * 16 + hh * 8);
        float4 u0 = zp[0], u1 = zp[1];
        u0.x += dt * (g[0] + sB2[base + 0]);
        u0.y += dt * (g[1] + sB2[base + 1]);
        u0.z += dt * (g[2] + sB2[base + 2]);
        u0.w += dt * (g[3] + sB2[base + 3]);
        u1.x += dt * (g[4] + sB2[base + 4]);
        u1.y += dt * (g[5] + sB2[base + 5]);
        u1.z += dt * (g[6] + sB2[base + 6]);
        u1.w += dt * (g[7] + sB2[base + 7]);
        zp[0] = u0; zp[1] = u1;
      }
    }
  }

  asm volatile("" ::: "memory");

  // Write final state back to global.
#pragma unroll
  for (int p = 0; p < 8; ++p) {
    const float4* lsrc = (const float4*)(zw + ((p * 2 + hh) * 16 + nn) * 16);
    float4* gdst = (float4*)(zout + (size_t)j * DN + p * 32 + hh * 16);
#pragma unroll
    for (int q = 0; q < 4; ++q) gdst[q] = lsrc[q];
  }
}

extern "C" void kernel_launch(void* const* d_in, const int* in_sizes, int n_in,
                              void* d_out, int out_size, void* d_ws,
                              size_t ws_size, hipStream_t stream) {
  const float* z0 = (const float*)d_in[0];
  const float* tr = (const float*)d_in[1];
  const float* W1 = (const float*)d_in[2];
  const float* b1 = (const float*)d_in[3];
  const float* wt = (const float*)d_in[4];
  const float* W2 = (const float*)d_in[5];
  const float* b2 = (const float*)d_in[6];
  __bf16* wsW = (__bf16*)d_ws;  // 131072 bf16 halves = 256 KB packed weights
  float* out = (float*)d_out;

  // One-time repack of W1^T/W2^T into WMMA A-fragment order.
  pack_weights_kernel<<<dim3(512), dim3(256), 0, stream>>>(W1, W2, wsW);

  // 128 blocks x 8 waves x 16 batch-cols = 16384 rows.
  const size_t smem_bytes = 262144 + 3 * 256 * sizeof(float);  // 265216 B
  ode_fused_kernel<<<dim3(BN / 128), dim3(256), smem_bytes, stream>>>(
      z0, tr, b1, wt, b2, wsW, out);
}
